// ExtractorLoss_13331578487304
// MI455X (gfx1250) — matrix operand order repaired and checked by reading, not compiled, not run on hardware
//
#include <hip/hip_runtime.h>
#include <hip/hip_fp16.h>
#include <math.h>

// CDNA5 / gfx1250 WMMA types: 16 halfs = 8 VGPRs (A/B), 8 floats = 8 VGPRs (C/D)
typedef __attribute__((ext_vector_type(16))) _Float16 v16h;
typedef __attribute__((ext_vector_type(8)))  float    v8f;

#define WAVES_PER_BLOCK 8
#define THREADS (WAVES_PER_BLOCK * 32)
#define NPAD 2048   // padded signal length in LDS (covers N up to 2048; N=900 here)

// -------------------------------------------------------------------------
// Kernel 1: one workgroup per batch element. Computes per-batch SNR (dB).
//
// PSD factorization: n = 32q + r.
//   P[f,q] = sum_r cos(w_f r) x[32q+r],  S[f,q] = sum_r sin(w_f r) x[32q+r]
//   -> two 16x16x32 f16 WMMAs per (freq-tile, q-tile), f32 accumulate.
//   c[f] = sum_q cosA*P - sinA*S ; s[f] = sum_q sinA*P + cosA*S,
//   with A = w_f * 32q. power = c^2 + s^2 (exact angle-addition identity).
//
// A-matrix cos/sin tables are generated with a complex-rotation recurrence
// (full-rate FMAs) seeded by 3 hardware __sincosf calls per lane per tile;
// the per-(row,q) outer rotation uses __sincosf (hardware v_sin/v_cos).
// Max |angle| ~ 2*pi*3*900/30 ~ 565 rad: HW reduction phase error
// ~ arg*2^-24 ~ 3e-5 rad, far below the f16 GEMM quantization noise.
// -------------------------------------------------------------------------
__global__ __launch_bounds__(THREADS) void extractor_psd_kernel(
    const float* __restrict__ x,       // [B,N]
    const float* __restrict__ f_true,  // [B]
    const float* __restrict__ fs,      // [B]
    const float* __restrict__ p_delta,
    const float* __restrict__ p_s,
    const float* __restrict__ p_fmin,
    const float* __restrict__ p_fmax,
    float* __restrict__ snr_out,       // [B] (workspace)
    int N)
{
    const int b    = blockIdx.x;
    const int tid  = threadIdx.x;
    const int lane = tid & 31;
    const int wave = tid >> 5;
    const int nlo  = lane & 15;  // column / row-within-half index
    const int khi  = lane >> 4;  // which lane-half (selects K sub-range)

    __shared__ _Float16 xs[NPAD];
    __shared__ float sWsh[WAVES_PER_BLOCK], sUsh[WAVES_PER_BLOCK];
    __shared__ unsigned cUsh[WAVES_PER_BLOCK];

    const float delta = p_delta[0];
    const float sfrq  = p_s[0];
    const float fmin  = p_fmin[0];
    const float fmax  = p_fmax[0];
    const float ft    = f_true[b];
    const float fsb   = fs[b];
    const float winv  = 6.283185307179586f / fsb;   // 2*pi/fs

    // Replicate numpy float arange length semantics in double precision.
    // Fw = len(arange(-delta, delta+s, s)); K = ceil((fmax-fmin)/s)+2.
    // (Overestimating K by a ULP-induced +1 is harmless: extra bins fail both masks.)
    const int Fw  = (int)ceil(((double)delta + (double)sfrq + (double)delta) / (double)sfrq);
    const int K   = (int)ceil(((double)fmax - (double)fmin) / (double)sfrq) + 2;
    const int Ftot   = Fw + 2 * K;
    const int ntiles = (Ftot + 15) >> 4;

    // q-tiles: q in [0, ceil(N/32)), grouped by 16; padded LDS makes tails exact zeros.
    int QT = (N + (32 * 16 - 1)) / (32 * 16);
    if (QT > NPAD / (32 * 16)) QT = NPAD / (32 * 16);

    // Stage signal into LDS as f16, zero-padded.
    for (int i = tid; i < NPAD; i += THREADS) {
        float v = (i < N) ? x[(size_t)b * N + i] : 0.0f;
        xs[i] = (_Float16)v;
    }
    __syncthreads();

    // frequency of global bin j (matches reference's f32 arithmetic & op order)
    auto freq_of = [&](int j) -> float {
        if (j < Fw) {
            float off = (float)((double)(-delta) + (double)j * (double)sfrq); // np.arange elt, f64->f32
            return ft + off;
        } else if (j < Fw + K) {
            return fmin + (float)(j - Fw) * sfrq;                 // f_min + k*s
        } else {
            return ((ft + delta) + sfrq) + (float)(j - Fw - K) * sfrq; // f_true+delta+s+k*s
        }
    };

    float accW = 0.0f, accU = 0.0f;
    unsigned accC = 0;

    for (int t = wave; t < ntiles; t += WAVES_PER_BLOCK) {
        // ---- A matrices (cos / sin), 16x32 f16 ----
        // ISA layout: lane holds row M = lane&15; VGPR v holds
        // K0 = (v>>2)*16 + (lane>>4)*8 + (v&3)*2 and K0+1 (packed lo/hi):
        // i.e. element e<8  <-> K = khi*8 + e,
        //      element e>=8 <-> K = 16 + khi*8 + (e-8)   (consecutive K runs).
        const float wA = winv * freq_of(t * 16 + nlo);

        // Seeds: start angle, unit step, 16-step jump (3 HW sincos).
        float cstep, sstep, cst, sst, cj, sj;
        __sincosf(wA,                      &sstep, &cstep);
        __sincosf(wA * (float)(khi << 3),  &sst,   &cst);
        __sincosf(wA * 16.0f,              &sj,    &cj);

        v16h Ac, As;
        {
            float c0 = cst, s0 = sst;
            #pragma unroll
            for (int e = 0; e < 8; ++e) {           // K = khi*8 + e
                Ac[e] = (_Float16)c0;  As[e] = (_Float16)s0;
                const float cn = c0 * cstep - s0 * sstep;
                const float sn = s0 * cstep + c0 * sstep;
                c0 = cn; s0 = sn;
            }
            c0 = cst * cj - sst * sj;               // rotate start by 16 steps
            s0 = sst * cj + cst * sj;
            #pragma unroll
            for (int e = 8; e < 16; ++e) {          // K = 16 + khi*8 + (e-8)
                Ac[e] = (_Float16)c0;  As[e] = (_Float16)s0;
                const float cn = c0 * cstep - s0 * sstep;
                const float sn = s0 * cstep + c0 * sstep;
                c0 = cn; s0 = sn;
            }
        }

        // Per-row angular rates for the outer rotation (hoisted out of q loop).
        // D layout: element v <-> row M = v + 8*(lane>>4), col N = lane&15.
        float wrow[8];
        #pragma unroll
        for (int v = 0; v < 8; ++v)
            wrow[v] = winv * freq_of(t * 16 + v + (khi << 3));

        float cacc[8] = {0, 0, 0, 0, 0, 0, 0, 0};
        float sacc[8] = {0, 0, 0, 0, 0, 0, 0, 0};

        for (int qt = 0; qt < QT; ++qt) {
            // ---- B matrix (32x16 f16) from LDS ----
            // ISA layout: lane holds column N = lane&15; element e<8 <-> K = khi*8 + e,
            // element e>=8 <-> K = 16 + khi*8 + (e-8) (same K order as A).
            const int q  = qt * 16 + nlo;
            const int n0 = q * 32 + (khi << 3);
            v16h Bx;
            #pragma unroll
            for (int e = 0; e < 16; ++e) Bx[e] = xs[n0 + e + ((e >> 3) << 3)];

            const v8f zacc = {0.f, 0.f, 0.f, 0.f, 0.f, 0.f, 0.f, 0.f};
            // D = A x B + 0 : f32 16x16 accumulator
            v8f P = __builtin_amdgcn_wmma_f32_16x16x32_f16(
                        false, Ac, false, Bx, (short)0, zacc, false, false);
            v8f S = __builtin_amdgcn_wmma_f32_16x16x32_f16(
                        false, As, false, Bx, (short)0, zacc, false, false);

            // Combine with outer rotation at angle w_row * 32q.
            const float qang = (float)(q << 5);
            #pragma unroll
            for (int v = 0; v < 8; ++v) {
                float sa, ca;
                __sincosf(wrow[v] * qang, &sa, &ca);
                cacc[v] += ca * P[v] - sa * S[v];
                sacc[v] += sa * P[v] + ca * S[v];
            }
        }

        // Butterfly-sum over the 16 q-columns (stays within each lane half).
        #pragma unroll
        for (int v = 0; v < 8; ++v) {
            float cv = cacc[v], sv = sacc[v];
            #pragma unroll
            for (int m = 1; m < 16; m <<= 1) {
                cv += __shfl_xor(cv, m, 32);
                sv += __shfl_xor(sv, m, 32);
            }
            cacc[v] = cv; sacc[v] = sv;
        }

        // Lanes 0 and 16 own rows 0-7 / 8-15 of this tile: classify & accumulate.
        if (nlo == 0) {
            #pragma unroll
            for (int v = 0; v < 8; ++v) {
                const int j = t * 16 + v + (khi << 3);
                if (j >= Ftot) continue;
                const float p = cacc[v] * cacc[v] + sacc[v] * sacc[v];
                if (j < Fw) {
                    accW += p;
                } else if (j < Fw + K) {
                    if (freq_of(j) < ft - delta) { accU += p; accC++; }
                } else {
                    if (freq_of(j) < fmax + sfrq) { accU += p; accC++; }
                }
            }
        }
    }

    // Merge lane 0 <- lane 16, then fixed-order cross-wave reduce (deterministic).
    accW += __shfl_xor(accW, 16, 32);
    accU += __shfl_xor(accU, 16, 32);
    accC += __shfl_xor(accC, 16, 32);
    if (lane == 0) { sWsh[wave] = accW; sUsh[wave] = accU; cUsh[wave] = accC; }
    __syncthreads();
    if (tid == 0) {
        float tW = 0.f, tU = 0.f; unsigned tC = 0;
        for (int w = 0; w < WAVES_PER_BLOCK; ++w) { tW += sWsh[w]; tU += sUsh[w]; tC += cUsh[w]; }
        const float term1 = tW / (float)Fw;
        const float term2 = tU / (float)tC;
        snr_out[b] = 10.0f * log10f(term1 / term2);
    }
}

// -------------------------------------------------------------------------
// Kernel 2: loss = -mean_b(snr_b)
// -------------------------------------------------------------------------
__global__ __launch_bounds__(256) void extractor_reduce_kernel(
    const float* __restrict__ snr, float* __restrict__ out, int B)
{
    __shared__ float sh[256];
    const int tid = threadIdx.x;
    float v = 0.f;
    for (int i = tid; i < B; i += 256) v += snr[i];
    sh[tid] = v;
    __syncthreads();
    for (int s = 128; s > 0; s >>= 1) {
        if (tid < s) sh[tid] += sh[tid + s];
        __syncthreads();
    }
    if (tid == 0) out[0] = -sh[0] / (float)B;
}

extern "C" void kernel_launch(void* const* d_in, const int* in_sizes, int n_in,
                              void* d_out, int out_size, void* d_ws, size_t ws_size,
                              hipStream_t stream) {
    const float* x          = (const float*)d_in[0];
    const float* f_true     = (const float*)d_in[1];
    const float* fs         = (const float*)d_in[2];
    const float* delta      = (const float*)d_in[3];
    const float* sampling_f = (const float*)d_in[4];
    const float* f_min      = (const float*)d_in[5];
    const float* f_max      = (const float*)d_in[6];

    const int B = in_sizes[1];
    const int N = in_sizes[0] / B;

    float* snr = (float*)d_ws;  // B floats of scratch; fully rewritten every call

    extractor_psd_kernel<<<B, THREADS, 0, stream>>>(
        x, f_true, fs, delta, sampling_f, f_min, f_max, snr, N);
    extractor_reduce_kernel<<<1, 256, 0, stream>>>(snr, (float*)d_out, B);
}
